// AssignAttention_53412213293476
// MI455X (gfx1250) — matrix operand order repaired and checked by reading, not compiled, not run on hardware
//
#include <hip/hip_runtime.h>
#include <hip/hip_bf16.h>

// ---------------------------------------------------------------------------
// AssignAttention on MI455X (gfx1250): bf16 WMMA, fused chunked attention.
// B=16, N=256, S=4096, C=512, H=8, hd=64. Softmax over the QUERY axis (n).
// Round 5: TDM staging via the 6-arg tensor_load_to_lds builtin (this
// toolchain is the clang-23/therock flavor), extra descriptor group zeroed.
// ---------------------------------------------------------------------------

typedef __attribute__((ext_vector_type(16))) __bf16 v16bf;
typedef __attribute__((ext_vector_type(8)))  float  v8f;
typedef __attribute__((ext_vector_type(4)))  unsigned uint32x4;
typedef __attribute__((ext_vector_type(8)))  int      int32x8;
typedef __attribute__((ext_vector_type(4)))  int      int32x4;

__device__ __forceinline__ unsigned bfbits(float f) {
  unsigned u = __builtin_bit_cast(unsigned, f);
  return (u + 0x7FFFu + ((u >> 16) & 1u)) >> 16;   // RNE
}
__device__ __forceinline__ __bf16 f2bf(float f) {
  unsigned short s = (unsigned short)bfbits(f);
  return __builtin_bit_cast(__bf16, s);
}

__device__ __forceinline__ v8f wmma_bf16(v16bf a, v16bf b, v8f c) {
  return __builtin_amdgcn_wmma_f32_16x16x32_bf16(
      false, a, false, b, (short)0, c, false, false);
}

// ---------------------------------------------------------------------------
// TDM: load a 2D tile of bf16 (tile_d0 elems/row, tile_d1 rows, row stride
// stride_elems) from global into packed LDS. D# per CDNA5 ISA ch.8:
//  group0: [1:0]=count=1 | [63:32]=lds_addr | [120:64]=global_addr | type=2
//  group1: data_size(=1 ->2B) | tensor_dim0/1 (tight) | tile_dim0/1 | stride0
//  groups 2+: zero (tile_dim2=0 -> dims 2..4 unused)
// ---------------------------------------------------------------------------
__device__ __forceinline__ void tdm_load_2d(const void* gptr, void* sptr,
                                            unsigned tile_d0, unsigned tile_d1,
                                            unsigned stride_elems) {
  const unsigned long long ga = (unsigned long long)(uintptr_t)gptr;
  const unsigned lds = (unsigned)(uintptr_t)sptr;   // flat-shared low 32 = LDS addr
  uint32x4 g0;
  g0[0] = 1u;                                                   // count=1
  g0[1] = lds;                                                  // lds_addr (bytes)
  g0[2] = (unsigned)ga;                                         // global_addr[31:0]
  g0[3] = (unsigned)((ga >> 32) & 0x01FFFFFFu) | (2u << 30);    // [56:32] | type=2
  int32x8 g1;
  g1[0] = (int)(1u << 16);                                      // data_size=1 (2B)
  g1[1] = (int)((tile_d0 & 0xFFFFu) << 16);                     // tensor_dim0 lo16
  g1[2] = (int)((tile_d0 >> 16) | ((tile_d1 & 0xFFFFu) << 16)); // td0 hi | td1 lo
  g1[3] = (int)((tile_d1 >> 16) | ((tile_d0 & 0xFFFFu) << 16)); // td1 hi | tile_dim0
  g1[4] = (int)(tile_d1 & 0xFFFFu);                             // tile_dim1, tile_dim2=0
  g1[5] = (int)stride_elems;                                    // dim0 stride lo32
  g1[6] = 0;                                                    // stride0 hi | stride1 lo
  g1[7] = 0;
  const int32x4 gz4 = {0, 0, 0, 0};
  const int32x8 gz8 = {0, 0, 0, 0, 0, 0, 0, 0};
  __builtin_amdgcn_tensor_load_to_lds(g0, g1, gz4, gz4, gz8, 0);
}
__device__ __forceinline__ void tdm_wait0() {
  __builtin_amdgcn_s_wait_tensorcnt(0);
}

// A-matrix 16x32 bf16 fragment: lane m=L&15, half=L>>4;
// slot i in 0..15 -> k = 16*half + i
template<int LD>
__device__ __forceinline__ v16bf load_a_frag(const __bf16* base, int lane) {
  const int m = lane & 15, half = lane >> 4;
  const __bf16* p = base + m * LD;
  v16bf a;
#pragma unroll
  for (int i = 0; i < 16; ++i) a[i] = p[16 * half + i];
  return a;
}

// B-matrix 32x16 bf16 fragment: lane n=L&15, half=L>>4; slot i -> k=16*half+i.
// Element (k,n) at base[k + n*SN] (k contiguous -> ds_load_b128 pairs).
template<int SN>
__device__ __forceinline__ v16bf load_b_fragk(const __bf16* base, int lane) {
  const int n = lane & 15, half = lane >> 4;
  const __bf16* p = base + n * SN + 16 * half;
  v16bf b;
#pragma unroll
  for (int i = 0; i < 16; ++i) b[i] = p[i];
  return b;
}

// ---------------------------------------------------------------------------
// Prep 1: packed f32 -> bf16 elementwise (8 elems/thread).
// ---------------------------------------------------------------------------
__global__ __launch_bounds__(256) void cvt_f32_bf16(
    const float* __restrict__ src, __bf16* __restrict__ dst, int n8) {
  const int i = blockIdx.x * 256 + threadIdx.x;
  if (i >= n8) return;
  const float4 f0 = *(const float4*)(src + (size_t)i * 8);
  const float4 f1 = *(const float4*)(src + (size_t)i * 8 + 4);
  uint4 o;
  o.x = bfbits(f0.x) | (bfbits(f0.y) << 16);
  o.y = bfbits(f0.z) | (bfbits(f0.w) << 16);
  o.z = bfbits(f1.x) | (bfbits(f1.y) << 16);
  o.w = bfbits(f1.z) | (bfbits(f1.w) << 16);
  *(uint4*)(dst + (size_t)i * 8) = o;
}

// ---------------------------------------------------------------------------
// Prep 2: WT[n][k] bf16 = W[k][n] f32, 512x512 (tiny; coalesced reads).
// ---------------------------------------------------------------------------
__global__ __launch_bounds__(256) void wtrans_bf16(
    const float* __restrict__ W, __bf16* __restrict__ WT) {
  const int id = blockIdx.x * 256 + threadIdx.x;   // 0..262143
  const int k = id >> 9, n = id & 511;
  WT[(size_t)n * 512 + k] = f2bf(W[(size_t)k * 512 + n]);
}

// ---------------------------------------------------------------------------
// Prep 3: kT[(b*8+h)*64 + c][s] = kbf[(b*4096+s)*512 + h*64 + c]
// LDS-tiled 64x64 transpose. grid (128, 64).
// ---------------------------------------------------------------------------
__global__ __launch_bounds__(256) void ktrans_bf16(
    const __bf16* __restrict__ kbf, __bf16* __restrict__ kT) {
  __shared__ __bf16 tile[64 * 72];   // stride 72 elems = 144 B (16B aligned)
  const int bh = blockIdx.x;         // b*8+h
  const int st = blockIdx.y;         // s tile
  const int b = bh >> 3, h = bh & 7;
  const int tid = threadIdx.x;
  const size_t src = ((size_t)b * 4096 + st * 64) * 512 + h * 64;

  for (int e = tid; e < 512; e += 256) {
    const int row = e >> 3, c8 = (e & 7) * 8;   // row = s, c8 = c
    *(uint4*)&tile[row * 72 + c8] = *(const uint4*)&kbf[src + (size_t)row * 512 + c8];
  }
  __syncthreads();

  const int c = tid >> 2, sseg = (tid & 3) * 16;  // 16 s per thread
  unsigned w[8];
#pragma unroll
  for (int j = 0; j < 8; ++j) {
    const int s = sseg + j * 2;
    const unsigned lo = __builtin_bit_cast(unsigned short, tile[s * 72 + c]);
    const unsigned hi = __builtin_bit_cast(unsigned short, tile[(s + 1) * 72 + c]);
    w[j] = lo | (hi << 16);
  }
  __bf16* dst = &kT[((size_t)bh * 64 + c) * 4096 + st * 64 + sseg];
  uint4 o0; o0.x = w[0]; o0.y = w[1]; o0.z = w[2]; o0.w = w[3];
  uint4 o1; o1.x = w[4]; o1.y = w[5]; o1.z = w[6]; o1.w = w[7];
  *(uint4*)dst = o0;
  *(uint4*)(dst + 8) = o1;
}

// ---------------------------------------------------------------------------
// Kernel: O[bf16, M x 512] = Abf[bf16, M x 512] @ WT' (WT is [n][k] bf16)
// Block tile 128x64, K chunks of 32; staging via TDM (wave 0 issues).
// ---------------------------------------------------------------------------
__global__ __launch_bounds__(256) void proj_gemm_bf16(
    const __bf16* __restrict__ A, const __bf16* __restrict__ WT,
    __bf16* __restrict__ O) {
  __shared__ __bf16 a_lds[128][32];
  __shared__ __bf16 wT[64][32];      // wT[n][k]

  const int tid  = threadIdx.x;
  const int lane = tid & 31, wv = tid >> 5;
  const int half = lane >> 4, lanen = lane & 15;
  const int mblock = blockIdx.x * 128;
  const int nblock = blockIdx.y * 64;
  const int mrow = (wv & 3) * 32;
  const int ncol = (wv >> 2) * 32;

  v8f acc[2][2];
#pragma unroll
  for (int mt = 0; mt < 2; ++mt)
#pragma unroll
    for (int nt = 0; nt < 2; ++nt) acc[mt][nt] = (v8f){};

  for (int kc = 0; kc < 16; ++kc) {
    const int koff = kc * 32;
    if (wv == 0) {   // one wave drives the DMA
      tdm_load_2d(&A[(size_t)mblock * 512 + koff], &a_lds[0][0], 32, 128, 512);
      tdm_load_2d(&WT[(size_t)nblock * 512 + koff], &wT[0][0], 32, 64, 512);
      tdm_wait0();
    }
    __syncthreads();

    v16bf af0 = load_a_frag<32>(&a_lds[mrow][0], lane);
    v16bf af1 = load_a_frag<32>(&a_lds[mrow + 16][0], lane);
#pragma unroll
    for (int nt = 0; nt < 2; ++nt) {
      v16bf bf = load_b_fragk<32>(&wT[ncol + nt * 16][0], lane);
      acc[0][nt] = wmma_bf16(af0, bf, acc[0][nt]);
      acc[1][nt] = wmma_bf16(af1, bf, acc[1][nt]);
    }
    __syncthreads();
  }

#pragma unroll
  for (int mt = 0; mt < 2; ++mt)
#pragma unroll
    for (int nt = 0; nt < 2; ++nt)
#pragma unroll
      for (int v = 0; v < 8; ++v) {
        const int gm = mblock + mrow + mt * 16 + v + 8 * half;
        const int gn = nblock + ncol + nt * 16 + lanen;
        O[(size_t)gm * 512 + gn] = f2bf(acc[mt][nt][v]);
      }
}

// ---------------------------------------------------------------------------
// Fused attention per (b,h). 256 threads = 8 waves, S chunks of 64.
// LDS map (dynamic):
//   q_lds  bf16[256*64]   @      0  (32768 B)
//   sc_lds f32 [256*64]   @  32768  (65536 B)
//   at_lds bf16[256*64]   @  98304  (32768 B)
//   k_lds  bf16[ 64*64]   @ 131072  ( 8192 B)  [s][c]
//   vT     bf16[ 64*64]   @ 139264  ( 8192 B)  [c][s]  (from kT, TDM)
//   red    f32 [64*4]     @ 147456  ( 1024 B)
//   cinv   f32 [64]       @ 148480  (  256 B)
//   cmax   f32 [64]       @ 148736  (  256 B)
//   rsum   f32 [256]      @ 148992  ( 1024 B)   total 150016 B
// ---------------------------------------------------------------------------
#define ATTN_SMEM_BYTES 150016

__global__ __launch_bounds__(256) void assign_attn_fused(
    const __bf16* __restrict__ qp, const __bf16* __restrict__ kp,
    const __bf16* __restrict__ kT, float* __restrict__ out) {
  extern __shared__ char smem[];
  __bf16* q_lds  = (__bf16*)(smem);
  float*  sc_lds = (float*)(smem + 32768);
  __bf16* at_lds = (__bf16*)(smem + 98304);
  __bf16* k_lds  = (__bf16*)(smem + 131072);
  __bf16* vT     = (__bf16*)(smem + 139264);
  float*  red    = (float*)(smem + 147456);
  float*  cinv   = (float*)(smem + 148480);
  float*  cmax   = (float*)(smem + 148736);
  float*  rsum   = (float*)(smem + 148992);

  const int tid  = threadIdx.x;
  const int lane = tid & 31, wv = tid >> 5;
  const int half = lane >> 4, lanen = lane & 15;
  const int b = blockIdx.x >> 3, h = blockIdx.x & 7;   // H = 8
  const size_t qbase = ((size_t)b * 256) * 512 + h * 64;
  const size_t kbase = ((size_t)b * 4096) * 512 + h * 64;
  const size_t vbase = (size_t)blockIdx.x * 64 * 4096;  // kT[(b*8+h)][c][s]
  const int mrow = wv * 32;

  // TDM-stage resident q tile [256 x 64] bf16
  if (wv == 0) {
    tdm_load_2d(&qp[qbase], q_lds, 64, 256, 512);
    tdm_wait0();
  }

  v8f accp[2][4];
#pragma unroll
  for (int mt = 0; mt < 2; ++mt)
#pragma unroll
    for (int nt = 0; nt < 4; ++nt) accp[mt][nt] = (v8f){};
  float rs = 0.f;
  __syncthreads();

  for (int ch = 0; ch < 64; ++ch) {
    const int s0 = ch * 64;
    __syncthreads();  // previous chunk's GEMM done before restaging k/v

    if (wv == 0) {    // TDM-stage k chunk [64 s][64 c] and vT chunk [64 c][64 s]
      tdm_load_2d(&kp[kbase + (size_t)s0 * 512], k_lds, 64, 64, 512);
      tdm_load_2d(&kT[vbase + s0], vT, 64, 64, 4096);
      tdm_wait0();
    }
    if (ch + 1 < 64) {  // global_prefetch of next chunk's lines
      __builtin_prefetch(&kp[kbase + (size_t)(s0 + 64 + (tid & 63)) * 512], 0, 1);
      __builtin_prefetch(&kT[vbase + (size_t)(tid & 63) * 4096 + s0 + 64], 0, 1);
    }
    __syncthreads();

    // scores: wave rows [mrow, mrow+32) x cols [0,64), K = 64 (2 steps of 32)
    v8f accs[2][4];
#pragma unroll
    for (int mt = 0; mt < 2; ++mt)
#pragma unroll
      for (int nt = 0; nt < 4; ++nt) accs[mt][nt] = (v8f){};
#pragma unroll
    for (int ks = 0; ks < 2; ++ks) {
      v16bf a0 = load_a_frag<64>(&q_lds[mrow * 64 + ks * 32], lane);
      v16bf a1 = load_a_frag<64>(&q_lds[(mrow + 16) * 64 + ks * 32], lane);
#pragma unroll
      for (int nt = 0; nt < 4; ++nt) {
        // B element (k=c, n=s) = k_lds[(nt*16 + n)*64 + ks*32 + k]  (k contig)
        v16bf bb = load_b_fragk<64>(&k_lds[(nt * 16) * 64 + ks * 32], lane);
        accs[0][nt] = wmma_bf16(a0, bb, accs[0][nt]);
        accs[1][nt] = wmma_bf16(a1, bb, accs[1][nt]);
      }
    }
    // scatter scaled scores (scale = hd^-0.5 = 0.125)
#pragma unroll
    for (int mt = 0; mt < 2; ++mt)
#pragma unroll
      for (int nt = 0; nt < 4; ++nt)
#pragma unroll
        for (int v = 0; v < 8; ++v) {
          const int m = mrow + mt * 16 + v + 8 * half;
          const int c = nt * 16 + lanen;
          sc_lds[m * 64 + c] = accs[mt][nt][v] * 0.125f;
        }
    __syncthreads();

    // column softmax over n (exact: all 256 rows resident). 4 threads/col.
    {
      const int col = tid & 63, qd = tid >> 6;
      float mx = -3.4e38f;
      for (int r = qd * 64; r < qd * 64 + 64; ++r)
        mx = fmaxf(mx, sc_lds[r * 64 + col]);
      red[col * 4 + qd] = mx;
    }
    __syncthreads();
    if (tid < 64)
      cmax[tid] = fmaxf(fmaxf(red[tid * 4], red[tid * 4 + 1]),
                        fmaxf(red[tid * 4 + 2], red[tid * 4 + 3]));
    __syncthreads();
    {
      const int col = tid & 63, qd = tid >> 6;
      const float cm = cmax[col];
      float sm = 0.f;
      for (int r = qd * 64; r < qd * 64 + 64; ++r) {
        const float e = __expf(sc_lds[r * 64 + col] - cm);
        sc_lds[r * 64 + col] = e;
        sm += e;
      }
      red[col * 4 + qd] = sm;
    }
    __syncthreads();
    if (tid < 64)
      cinv[tid] = 1.0f / (red[tid * 4] + red[tid * 4 + 1] +
                          red[tid * 4 + 2] + red[tid * 4 + 3]);
    __syncthreads();

    // normalize (thread tid owns row tid), running rowsum, bf16 pack (vec4)
    {
      const int row = tid;
#pragma unroll
      for (int c4 = 0; c4 < 16; ++c4) {
        const float4 sv = *(const float4*)&sc_lds[row * 64 + c4 * 4];
        const float4 iv = *(const float4*)&cinv[c4 * 4];
        const float a0 = sv.x * iv.x, a1 = sv.y * iv.y;
        const float a2 = sv.z * iv.z, a3 = sv.w * iv.w;
        rs += (a0 + a1) + (a2 + a3);
        uint2 pk;
        pk.x = bfbits(a0) | (bfbits(a1) << 16);
        pk.y = bfbits(a2) | (bfbits(a3) << 16);
        *(uint2*)&at_lds[row * 64 + c4 * 4] = pk;
      }
    }
    __syncthreads();

    // P += attn[256 x 64] @ v[64 x 64]  (B from transposed vT -> k contig)
#pragma unroll
    for (int ks = 0; ks < 2; ++ks) {
      v16bf a0 = load_a_frag<64>(&at_lds[mrow * 64 + ks * 32], lane);
      v16bf a1 = load_a_frag<64>(&at_lds[(mrow + 16) * 64 + ks * 32], lane);
#pragma unroll
      for (int nt = 0; nt < 4; ++nt) {
        // B element (k=s, n=c) = vT[(nt*16 + n)*64 + ks*32 + k]  (k contig)
        v16bf bb = load_b_fragk<64>(&vT[(nt * 16) * 64 + ks * 32], lane);
        accp[0][nt] = wmma_bf16(a0, bb, accp[0][nt]);
        accp[1][nt] = wmma_bf16(a1, bb, accp[1][nt]);
      }
    }
  }

  rsum[tid] = rs;
  __syncthreads();

  const size_t BNC = (size_t)16 * 256 * 512;
#pragma unroll
  for (int mt = 0; mt < 2; ++mt)
#pragma unroll
    for (int nt = 0; nt < 4; ++nt)
#pragma unroll
      for (int v = 0; v < 8; ++v) {
        const int m = mrow + mt * 16 + v + 8 * half;
        const int c = nt * 16 + lanen;
        const float d = fmaxf(rsum[m], 1.0f);       // clamp-normalize divisor
        const float val = accp[mt][nt][v] / d;
        const size_t idx = ((size_t)b * 256 + m) * 512 + h * 64 + c;
        out[idx] = val;          // out
        out[BNC + idx] = val;    // out_style (== out, value == key)
      }
}

// ---------------------------------------------------------------------------
extern "C" void kernel_launch(void* const* d_in, const int* in_sizes, int n_in,
                              void* d_out, int out_size, void* d_ws, size_t ws_size,
                              hipStream_t stream) {
  const float* query = (const float*)d_in[0];  // [16,256,512]
  const float* key   = (const float*)d_in[1];  // [16,4096,512]
  const float* Wq    = (const float*)d_in[2];  // [512,512]
  const float* Wk    = (const float*)d_in[3];  // [512,512]
  float* out = (float*)d_out;                  // [2, 16, 256, 512]

  char* ws = (char*)d_ws;
  __bf16* qbf = (__bf16*)(ws);                            //  4,194,304 B
  __bf16* kbf = (__bf16*)(ws + 4194304);                  // 67,108,864 B
  __bf16* kT  = (__bf16*)(ws + 71303168);                 // 67,108,864 B
  __bf16* WqT = (__bf16*)(ws + 138412032);                //    524,288 B
  __bf16* WkT = (__bf16*)(ws + 138936320);                //    524,288 B
  __bf16* qp  = (__bf16*)(ws + 139460608);                //  4,194,304 B
  __bf16* kp  = (__bf16*)(ws + 143654912);                // 67,108,864 B

  dim3 blk(256);
  // prep passes (one-shot, bandwidth-bound)
  cvt_f32_bf16<<<dim3(1024),  blk, 0, stream>>>(query, qbf, 262144);
  cvt_f32_bf16<<<dim3(16384), blk, 0, stream>>>(key,   kbf, 4194304);
  wtrans_bf16<<<dim3(1024), blk, 0, stream>>>(Wq, WqT);
  wtrans_bf16<<<dim3(1024), blk, 0, stream>>>(Wk, WkT);
  ktrans_bf16<<<dim3(128, 64), blk, 0, stream>>>(kbf, kT);

  // projections (bf16 WMMA, TDM-staged)
  proj_gemm_bf16<<<dim3(32, 8),  blk, 0, stream>>>(qbf, WqT, qp);
  proj_gemm_bf16<<<dim3(512, 8), blk, 0, stream>>>(kbf, WkT, kp);

  // fused attention
  (void)hipFuncSetAttribute((const void*)assign_attn_fused,
                            hipFuncAttributeMaxDynamicSharedMemorySize,
                            ATTN_SMEM_BYTES);
  assign_attn_fused<<<dim3(128), blk, ATTN_SMEM_BYTES, stream>>>(qp, kp, kT, out);
}
